// similarity_model_8375186227208
// MI455X (gfx1250) — matrix operand order repaired and checked by reading, not compiled, not run on hardware
//
#include <hip/hip_runtime.h>

// ---------------------------------------------------------------------------
// Word-similarity top-k for MI455X (gfx1250, wave32, WMMA).
//   q = emb[word_ids]           [512, 512]  f32 -> bf16
//   sim = q @ emb^T             [512, 500000]  (never materialized)
//   out = top-100 scores + ids per row (top-101 minus the self hit)
//
// Workspace layout (bytes):
//   [0,             524288)   qbf  : bf16 q matrix, 512x512
//   [524288,   25966592)      cand scores: [512][123][101] f32
//   [25966592, 51408896)      cand ids   : [512][123][101] i32
// ---------------------------------------------------------------------------

typedef __attribute__((ext_vector_type(16))) __bf16 v16bf;
typedef __attribute__((ext_vector_type(8)))  float  v8f;

constexpr int   V      = 500000;
constexpr int   D      = 512;
constexpr int   BATCH  = 512;
constexpr int   TOPK   = 100;
constexpr int   KEEP   = TOPK + 1;               // 101 (self hit + 100)
constexpr float NEG_BIG = -3.402823466e38f;

constexpr int BM     = 64;                       // batch rows per workgroup
constexpr int PANEL  = 128;                      // vocab rows staged per panel
constexpr int CHUNK  = 4096;                     // vocab rows per candidate chunk
constexpr int PANELS = CHUNK / PANEL;            // 32
constexpr int NCH    = (V + CHUNK - 1) / CHUNK;  // 123
constexpr int NCAND  = NCH * KEEP;               // 12423 candidates per row

constexpr int SA_LD  = D + 8;                    // 520 bf16 (bank-conflict pad)
constexpr int SB_LD  = PANEL + 8;                // 136 bf16

// LDS layout for the GEMM kernel (total 317,440 B of 320 KB)
constexpr size_t OFF_SA  = 0;                                    // 64x520 u16
constexpr size_t OFF_SBT = OFF_SA  + (size_t)BM * SA_LD * 2;     // 512x136 u16
constexpr size_t OFF_LS  = OFF_SBT + (size_t)D  * SB_LD * 2;     // 8x16x101 f32
constexpr size_t OFF_LI  = OFF_LS  + (size_t)8 * 16 * KEEP * 4;  // 8x16x101 i32
constexpr size_t OFF_SC  = OFF_LI  + (size_t)8 * 16 * KEEP * 4;  // 8x256 f32
constexpr size_t SMEM2   = OFF_SC  + (size_t)8 * 256 * 4;

// LDS layout for the merge kernel
constexpr size_t OFF3_CS = 0;
constexpr size_t OFF3_CI = (((size_t)NCAND * 4) + 15) & ~size_t(15);
constexpr size_t OFF3_RS = OFF3_CI + ((((size_t)NCAND * 4) + 15) & ~size_t(15));
constexpr size_t OFF3_RI = OFF3_RS + 256 * 4;
constexpr size_t SMEM3   = OFF3_RI + 256 * 4;

// Workspace offsets
constexpr size_t WS_QBF = 0;
constexpr size_t WS_CSC = WS_QBF + (size_t)BATCH * D * 2;
constexpr size_t WS_CID = WS_CSC + (size_t)BATCH * NCAND * 4;

__device__ __forceinline__ unsigned short f2bf(float f) {
  unsigned int u = __float_as_uint(f);
  unsigned int r = u + 0x7FFFu + ((u >> 16) & 1u);   // round-to-nearest-even
  if ((u & 0x7F800000u) == 0x7F800000u) r = u;       // inf/nan passthrough
  return (unsigned short)(r >> 16);
}

union FragBF {
  v16bf v;
  unsigned int u32[8];
};

// ---------------------------------------------------------------------------
// Kernel 1: gather query rows, convert to bf16.
// ---------------------------------------------------------------------------
__global__ void embed_gather_kernel(const int* __restrict__ word_ids,
                                    const float* __restrict__ emb,
                                    unsigned short* __restrict__ qbf) {
  const int b = blockIdx.x;
  int wid = word_ids[b];
  if (wid < 0) wid = 0;
  if (wid >= V) wid = V - 1;
  const float* src = emb + (size_t)wid * D;
  for (int d = threadIdx.x; d < D; d += blockDim.x)
    qbf[(size_t)b * D + d] = f2bf(src[d]);
}

// ---------------------------------------------------------------------------
// Kernel 2: bf16 WMMA GEMM fused with per-chunk top-101 selection.
// grid = (BATCH/BM, NCH) ; block = 256 (8 waves)
// Each wave: fixed 16-row M block, 4 concurrent 16-col N tiles (register
// blocked -> 4 independent WMMA accumulator chains, A fragment reused 4x).
// ---------------------------------------------------------------------------
__global__ void sim_gemm_topk_chunk_kernel(const float* __restrict__ emb,
                                           const unsigned short* __restrict__ qbf,
                                           float* __restrict__ cand_s,
                                           int*   __restrict__ cand_i) {
  extern __shared__ unsigned char smem[];
  unsigned short* sA   = (unsigned short*)(smem + OFF_SA);   // [BM][SA_LD]
  unsigned short* sBt  = (unsigned short*)(smem + OFF_SBT);  // [D][SB_LD] (transposed panel)
  float*          lsc  = (float*)(smem + OFF_LS);            // [8][16][KEEP]
  int*            lidv = (int*)(smem + OFF_LI);              // [8][16][KEEP]
  float*          sC   = (float*)(smem + OFF_SC);            // [8][16][16]

  const int tid  = threadIdx.x;
  const int wave = tid >> 5;
  const int lane = tid & 31;
  const int b0         = blockIdx.x * BM;
  const int chunk      = blockIdx.y;
  const int chunk_base = chunk * CHUNK;

  // ---- stage the q tile: 64 rows x 512 bf16, padded rows ----
  {
    const unsigned short* qg = qbf + (size_t)b0 * D;
    for (int i = tid * 8; i < BM * D; i += 256 * 8) {
      const int row = i / D, col = i % D;   // 8 contiguous share a row (512 % 8 == 0)
      *(uint4*)&sA[(size_t)row * SA_LD + col] = *(const uint4*)&qg[i];
    }
  }
  // ---- init the per-(wave,row) sorted lists ----
  for (int i = tid; i < 8 * 16 * KEEP; i += 256) { lsc[i] = NEG_BIG; lidv[i] = -1; }

  const int rb = wave & 3;           // row block within WG (rows rb*16 .. +16)
  const int nh = wave >> 2;          // which half of the 8 N-tiles per panel
  const int m0 = rb * 16;
  const int am = m0 + (lane & 15);   // A-fragment source row (ISA A layout)
  const int ah = lane >> 4;          // lane half selects K sub-runs
  const int nb = nh * 64;            // panel-local base column of this wave

  for (int p = 0; p < PANELS; ++p) {
    __syncthreads();   // previous panel fully consumed before restaging sBt
    // ---- stage panel: 128 vocab rows x 512 dims, f32 -> bf16, transposed ----
    {
      const int vr   = tid >> 1;       // 0..127 : panel-local vocab row
      const int dh   = tid & 1;        // half of the 512 dims
      const int vrow = chunk_base + p * PANEL + vr;
      unsigned short* dst = sBt + vr;
      if (vrow < V) {
        const float* src = emb + (size_t)vrow * D + dh * 256;
        if (vrow + PANEL < V)
          __builtin_prefetch(src + (size_t)PANEL * D, 0, 1);  // global_prefetch_b8
        for (int d = 0; d < 256; d += 4) {
          float4 f = *(const float4*)(src + d);
          const int dd = dh * 256 + d;
          dst[(size_t)(dd + 0) * SB_LD] = f2bf(f.x);
          dst[(size_t)(dd + 1) * SB_LD] = f2bf(f.y);
          dst[(size_t)(dd + 2) * SB_LD] = f2bf(f.z);
          dst[(size_t)(dd + 3) * SB_LD] = f2bf(f.w);
        }
      } else {
        for (int d = 0; d < 256; ++d) dst[(size_t)(dh * 256 + d) * SB_LD] = 0;
      }
    }
    __syncthreads();

    // ---- K loop: 4 concurrent N-tiles, 4 independent WMMA chains ----
    v8f acc0 = {0.f, 0.f, 0.f, 0.f, 0.f, 0.f, 0.f, 0.f};
    v8f acc1 = acc0, acc2 = acc0, acc3 = acc0;
    for (int ks = 0; ks < 16; ++ks) {
      const int k0 = ks * 32;
      FragBF a;
      // A 16x32 bf16: lane m holds K runs [hi*8,+8) and [16+hi*8,+8)
      *(uint4*)&a.u32[0] = *(const uint4*)&sA[(size_t)am * SA_LD + k0 + ah * 8];
      *(uint4*)&a.u32[4] = *(const uint4*)&sA[(size_t)am * SA_LD + k0 + 16 + ah * 8];
      // B 32x16 bf16: lane l holds column N = l%16, K runs [hi*8,+8) and [16+hi*8,+8)
      const int bn = nb + (lane & 15);
      FragBF bf0, bf1, bf2, bf3;
#pragma unroll
      for (int j = 0; j < 8; ++j) {
        const int bk = k0 + ((j < 4) ? (ah * 8 + 2 * j) : (16 + ah * 8 + 2 * (j - 4)));
        const unsigned short* brow = &sBt[(size_t)bk * SB_LD + bn];
        bf0.u32[j] = (unsigned int)brow[0]  | ((unsigned int)brow[SB_LD + 0]  << 16);
        bf1.u32[j] = (unsigned int)brow[16] | ((unsigned int)brow[SB_LD + 16] << 16);
        bf2.u32[j] = (unsigned int)brow[32] | ((unsigned int)brow[SB_LD + 32] << 16);
        bf3.u32[j] = (unsigned int)brow[48] | ((unsigned int)brow[SB_LD + 48] << 16);
      }
      acc0 = __builtin_amdgcn_wmma_f32_16x16x32_bf16(
          false, a.v, false, bf0.v, (short)0, acc0, false, false);
      acc1 = __builtin_amdgcn_wmma_f32_16x16x32_bf16(
          false, a.v, false, bf1.v, (short)0, acc1, false, false);
      acc2 = __builtin_amdgcn_wmma_f32_16x16x32_bf16(
          false, a.v, false, bf2.v, (short)0, acc2, false, false);
      acc3 = __builtin_amdgcn_wmma_f32_16x16x32_bf16(
          false, a.v, false, bf3.v, (short)0, acc3, false, false);
    }

    // ---- spill each tile to per-wave LDS, then per-row top-101 insert ----
    const int cn = lane & 15, ch = lane >> 4;
    float* cw = sC + wave * 256;
#pragma unroll
    for (int nt = 0; nt < 4; ++nt) {
      const v8f acc = (nt == 0) ? acc0 : (nt == 1) ? acc1 : (nt == 2) ? acc2 : acc3;
      // C layout: lane = N + 16*(M>=8), vgpr r -> M = r + 8*(lane>=16)
#pragma unroll
      for (int r = 0; r < 8; ++r) cw[(r + 8 * ch) * 16 + cn] = acc[r];
      asm volatile("s_wait_dscnt 0" ::: "memory");  // in-wave DS RAW
      if (lane < 16) {
        const int rr    = lane;
        const int lbase = (wave * 16 + rr) * KEEP;
        const float* cr = cw + rr * 16;
        const int vid0  = chunk_base + p * PANEL + nb + nt * 16;
        for (int c = 0; c < 16; ++c) {
          const int   vid = vid0 + c;
          const float s   = cr[c];
          if (vid < V && s > lsc[lbase + KEEP - 1]) {
            int pos = KEEP - 1;
            while (pos > 0 && lsc[lbase + pos - 1] < s) {
              lsc[lbase + pos]  = lsc[lbase + pos - 1];
              lidv[lbase + pos] = lidv[lbase + pos - 1];
              --pos;
            }
            lsc[lbase + pos]  = s;
            lidv[lbase + pos] = vid;
          }
        }
      }
      asm volatile("s_wait_dscnt 0" ::: "memory");  // list writes before sC reuse
    }
  }
  __syncthreads();

  // ---- merge the two wave lists per row, write chunk candidates ----
  if (tid < BM) {
    const int r  = tid;
    const int rr = r & 15;
    const int wA = (r >> 4);          // waves wA and wA+4 cover this row block
    const int baseA = (wA * 16 + rr) * KEEP;
    const int baseB = ((wA + 4) * 16 + rr) * KEEP;
    const size_t o  = ((size_t)(b0 + r) * NCH + chunk) * KEEP;
    int ia = 0, ib = 0;
    for (int k = 0; k < KEEP; ++k) {
      const float sa = lsc[baseA + ia], sb = lsc[baseB + ib];
      const bool ta = (sa >= sb);
      cand_s[o + k] = ta ? sa : sb;
      cand_i[o + k] = ta ? lidv[baseA + ia] : lidv[baseB + ib];
      if (ta) ++ia; else ++ib;
    }
  }
}

// ---------------------------------------------------------------------------
// Kernel 3: per batch row, exact top-101 of the 12,423 chunk candidates;
// drop rank 0 (self match), emit 100 scores + 100 ids (as float).
// ---------------------------------------------------------------------------
__global__ void topk_merge_kernel(const float* __restrict__ cand_s,
                                  const int*   __restrict__ cand_i,
                                  float* __restrict__ out) {
  extern __shared__ unsigned char smem[];
  float* cs = (float*)(smem + OFF3_CS);
  int*   ci = (int*)(smem + OFF3_CI);
  float* rs = (float*)(smem + OFF3_RS);
  int*   ri = (int*)(smem + OFF3_RI);

  const int b   = blockIdx.x;
  const int tid = threadIdx.x;
  const size_t base = (size_t)b * NCAND;
  for (int i = tid; i < NCAND; i += 256) { cs[i] = cand_s[base + i]; ci[i] = cand_i[base + i]; }
  __syncthreads();

  for (int j = 0; j < KEEP; ++j) {
    float bs = NEG_BIG;
    int   bi = NCAND;
    for (int i = tid; i < NCAND; i += 256) {
      const float s = cs[i];
      if (s > bs) { bs = s; bi = i; }   // keeps smallest index on ties
    }
    rs[tid] = bs; ri[tid] = bi;
    __syncthreads();
    for (int st = 128; st > 0; st >>= 1) {
      if (tid < st) {
        const float so = rs[tid + st];
        const int   io = ri[tid + st];
        if (so > rs[tid] || (so == rs[tid] && io < ri[tid])) { rs[tid] = so; ri[tid] = io; }
      }
      __syncthreads();
    }
    const int arg = ri[0];
    if (tid == 0) {
      if (j > 0) {
        out[(size_t)b * TOPK + (j - 1)] = rs[0];
        out[(size_t)BATCH * TOPK + (size_t)b * TOPK + (j - 1)] = (float)ci[arg];
      }
      cs[arg] = NEG_BIG;   // retire the winner
    }
    __syncthreads();
  }
}

// ---------------------------------------------------------------------------
extern "C" void kernel_launch(void* const* d_in, const int* in_sizes, int n_in,
                              void* d_out, int out_size, void* d_ws, size_t ws_size,
                              hipStream_t stream) {
  (void)in_sizes; (void)n_in; (void)out_size; (void)ws_size;
  const int*   word_ids = (const int*)d_in[0];
  const float* emb      = (const float*)d_in[1];
  // d_in[2] = topk scalar (100) -- fixed at compile time per reference.
  float* out = (float*)d_out;

  unsigned char* ws = (unsigned char*)d_ws;
  unsigned short* qbf = (unsigned short*)(ws + WS_QBF);
  float* cand_s       = (float*)(ws + WS_CSC);
  int*   cand_i       = (int*)(ws + WS_CID);

  embed_gather_kernel<<<BATCH, 256, 0, stream>>>(word_ids, emb, qbf);
  sim_gemm_topk_chunk_kernel<<<dim3(BATCH / BM, NCH), 256, SMEM2, stream>>>(
      emb, qbf, cand_s, cand_i);
  topk_merge_kernel<<<BATCH, 256, SMEM3, stream>>>(cand_s, cand_i, out);
}
